// PtrNet_11338713662134
// MI455X (gfx1250) — matrix operand order repaired and checked by reading, not compiled, not run on hardware
//
#include <hip/hip_runtime.h>
#include <hip/hip_bf16.h>

// ---------------------------------------------------------------------------
// PtrNet on MI455X (gfx1250): mixed-precision WMMA pipeline.
//  - all GEMMs (LSTM gates, refp projections, attention queries) use
//    v_wmma_f32_16x16x32_f16 with f32 accumulation
//  - weights cast to f16 once; [Wi|Wh] concatenated so each LSTM step is a
//    single K=512 GEMM on [x|h]
//  - refp_g/refp_p hoisted out of the decoder loop (single 65536x256x256 GEMM)
//  - attention: 1 block per batch row, wave32 shuffle reductions
//  - K-loop split into two branch-free phases (A1 then A2): no per-iteration
//    cndmask/select, pointer-increment addressing only
// ---------------------------------------------------------------------------

typedef __attribute__((ext_vector_type(16))) _Float16 v16h;
typedef __attribute__((ext_vector_type(8)))  _Float16 v8h;
typedef __attribute__((ext_vector_type(8)))  float    v8f;

#define Bn 512
#define Sn 128
#define En 256
#define Hn 256
#define H4n 1024

__device__ __forceinline__ float sigm(float x) { return 1.0f / (1.0f + __expf(-x)); }

// One K=32 step of the wave's 16x32 tile: A fragment in contiguous k order (k = 16h + s),
// two B fragments (N and N+16), two WMMAs.
__device__ __forceinline__ void wmma_step(const _Float16* __restrict__ ap, int kh,
                                          const _Float16* __restrict__ w0,
                                          const _Float16* __restrict__ w1,
                                          v8f& acc0, v8f& acc1)
{
  // lanes 0-15 hold K {0..15}; lanes 16-31 hold K {16..31}
  v8h alo = *(const v8h*)(ap + kh);
  v8h ahi = *(const v8h*)(ap + kh + 8);
  v16h a  = __builtin_shufflevector(alo, ahi, 0,1,2,3,4,5,6,7,8,9,10,11,12,13,14,15);
  v16h b0 = *(const v16h*)(w0);
  v16h b1 = *(const v16h*)(w1);
  __builtin_prefetch(w0 + 128, 0, 3);
  acc0 = __builtin_amdgcn_wmma_f32_16x16x32_f16(false, a, false, b0, (short)0, acc0, false, false);
  acc1 = __builtin_amdgcn_wmma_f32_16x16x32_f16(false, a, false, b1, (short)0, acc1, false, false);
}

// ---------------------------------------------------------------------------
// Generic WMMA GEMM:  C[M,N] = [A1 | A2] @ W^T + bias
//   A1: [M,K1] f16 row-major (lda1), A2: [M,K2] f16 (lda2, may be null/K2=0)
//   W : [N, K1+K2] f16 row-major (ldw)  == B^T, so lanes read contiguous K
//   out: f32 (Cf) and/or f16 (Ch), ldc
// Block = 256 threads = 8 waves; block tile 64(M) x 64(N); wave tile 16x32.
// Requires M%64==0, N%64==0, K1%32==0, K2%32==0 (true for all call sites).
// ---------------------------------------------------------------------------
__global__ __launch_bounds__(256)
void gemm_wmma(const _Float16* __restrict__ A1, int lda1, int K1,
               const _Float16* __restrict__ A2, int lda2, int K2,
               const _Float16* __restrict__ W,  int ldw,
               const float* __restrict__ bias,
               float* __restrict__ Cf, _Float16* __restrict__ Ch, int ldc)
{
  const int lane   = threadIdx.x & 31;
  const int wave   = threadIdx.x >> 5;   // 0..7
  const int wm     = wave >> 1;          // 0..3
  const int wn     = wave & 1;           // 0..1
  const int m0     = blockIdx.y * 64 + wm * 16;
  const int n0     = blockIdx.x * 64 + wn * 32;
  const int laneLo = lane & 15;
  const int hi     = lane >> 4;          // 0 or 1
  const int kh     = hi * 16;

  v8f acc0 = {}; v8f acc1 = {};

  const _Float16* ap = A1 + (size_t)(m0 + laneLo) * lda1;
  const _Float16* w0 = W + (size_t)(n0 + laneLo) * ldw + hi * 16;
  const _Float16* w1 = w0 + (size_t)16 * ldw;

  // phase 1: A1 columns (branch-free, pointer increments only)
#pragma unroll 4
  for (int k = 0; k < K1; k += 32) {
    wmma_step(ap, kh, w0, w1, acc0, acc1);
    ap += 32; w0 += 32; w1 += 32;
  }
  // phase 2: A2 columns (skipped when K2 == 0)
  if (K2 > 0) {
    const _Float16* ap2 = A2 + (size_t)(m0 + laneLo) * lda2;
#pragma unroll 4
    for (int k = 0; k < K2; k += 32) {
      wmma_step(ap2, kh, w0, w1, acc0, acc1);
      ap2 += 32; w0 += 32; w1 += 32;
    }
  }

  const int col0 = n0 + laneLo;
  const int col1 = col0 + 16;
  const float bv0 = bias ? bias[col0] : 0.0f;
  const float bv1 = bias ? bias[col1] : 0.0f;
  const int rowBase = m0 + hi * 8;
#pragma unroll
  for (int j = 0; j < 8; ++j) {
    const int row = rowBase + j;
    const float v0 = acc0[j] + bv0;
    const float v1 = acc1[j] + bv1;
    if (Cf) { Cf[(size_t)row * ldc + col0] = v0; Cf[(size_t)row * ldc + col1] = v1; }
    if (Ch) { Ch[(size_t)row * ldc + col0] = (_Float16)v0; Ch[(size_t)row * ldc + col1] = (_Float16)v1; }
  }
}

// ---------------------------------------------------------------------------
// embd[b,s,:] = f16(problems[b,s,0:4] @ embed_W + embed_b)   (K=4: VALU)
// ---------------------------------------------------------------------------
__global__ void embed_kernel(const float* __restrict__ problems,
                             const float* __restrict__ Wemb,
                             const float* __restrict__ bemb,
                             _Float16* __restrict__ embd, int n)
{
  int idx = blockIdx.x * blockDim.x + threadIdx.x;
  if (idx >= n) return;
  int e  = idx & (En - 1);
  int bs = idx >> 8;
  const float* p = problems + (size_t)bs * 4;
  float acc = bemb[e]
            + p[0] * Wemb[e] + p[1] * Wemb[En + e]
            + p[2] * Wemb[2 * En + e] + p[3] * Wemb[3 * En + e];
  embd[idx] = (_Float16)acc;
}

__global__ void cast_f16(const float* __restrict__ s, _Float16* __restrict__ d, int n)
{
  int i = blockIdx.x * blockDim.x + threadIdx.x;
  if (i < n) d[i] = (_Float16)s[i];
}

// Wcat[n, :] = [Wi[n,:K1] | Wh[n,:K2]] as f16
__global__ void cat_weights(const float* __restrict__ Wi, const float* __restrict__ Wh,
                            _Float16* __restrict__ Wc, int K1, int K2, int n)
{
  int idx = blockIdx.x * blockDim.x + threadIdx.x;
  if (idx >= n) return;
  int K = K1 + K2;
  int r = idx / K, k = idx - r * K;
  float v = (k < K1) ? Wi[(size_t)r * K1 + k] : Wh[(size_t)r * K2 + (k - K1)];
  Wc[idx] = (_Float16)v;
}

// LSTM pointwise gates (PyTorch order i,f,g,o). Updates c (f32), h (f16),
// optionally stores h into enc[b, t, :] for the encoder pass.
__global__ void lstm_gates(const float* __restrict__ g,
                           float* __restrict__ c, _Float16* __restrict__ h_h,
                           _Float16* __restrict__ enc_out, int t)
{
  int idx = blockIdx.x * blockDim.x + threadIdx.x;   // B*H
  if (idx >= Bn * Hn) return;
  int b = idx >> 8, d = idx & (Hn - 1);
  const float* gr = g + (size_t)b * H4n;
  float i  = sigm(gr[d]);
  float f  = sigm(gr[Hn + d]);
  float gg = tanhf(gr[2 * Hn + d]);
  float o  = sigm(gr[3 * Hn + d]);
  float cn = f * c[idx] + i * gg;
  float hn = o * tanhf(cn);
  c[idx]   = cn;
  h_h[idx] = (_Float16)hn;
  if (enc_out) enc_out[((size_t)b * Sn + t) * Hn + d] = (_Float16)hn;
}

// x[b,:] = (t==0) ? dec0 : embd[b, last[b], :]
__global__ void gather_x(const _Float16* __restrict__ embd,
                         const _Float16* __restrict__ dec0h,
                         const int* __restrict__ last,
                         _Float16* __restrict__ x, int t)
{
  int idx = blockIdx.x * blockDim.x + threadIdx.x;   // B*E
  if (idx >= Bn * En) return;
  int b = idx >> 8, e = idx & (En - 1);
  x[idx] = (t == 0) ? dec0h[e] : embd[((size_t)b * Sn + last[b]) * En + e];
}

// ---------------------------------------------------------------------------
// Glimpse attention (one block per batch row, 256 threads / 8 waves):
//   u[s] = 10 * sum_d V[d]*tanh(qp[d] + refp[b,s,d]), masked -1e8
//   q[b,:] = refp[b,:,:]^T @ softmax(u)  (written f16 for the next GEMM)
// ---------------------------------------------------------------------------
__global__ __launch_bounds__(256)
void glimpse_kernel(const float* __restrict__ qp, const _Float16* __restrict__ refp,
                    const float* __restrict__ V, const unsigned char* __restrict__ visited,
                    _Float16* __restrict__ q_out)
{
  const int b = blockIdx.x, tid = threadIdx.x;
  const int lane = tid & 31, wave = tid >> 5;
  __shared__ float qp_s[Hn], V_s[Hn], u_s[Sn], red_s[8];
  qp_s[tid] = qp[(size_t)b * Hn + tid];
  V_s[tid]  = V[tid];
  __syncthreads();
  const _Float16* rp = refp + (size_t)b * Sn * Hn;
  for (int s = wave; s < Sn; s += 8) {
    const _Float16* row = rp + (size_t)s * Hn;
    float sum = 0.0f;
    for (int d = lane; d < Hn; d += 32)
      sum += V_s[d] * tanhf(qp_s[d] + (float)row[d]);
    for (int off = 16; off > 0; off >>= 1) sum += __shfl_xor(sum, off, 32);
    if (lane == 0) u_s[s] = visited[(size_t)b * Sn + s] ? -1.0e8f : 10.0f * sum;
  }
  __syncthreads();
  float m = (tid < Sn) ? u_s[tid] : -3.4e38f;
  for (int off = 16; off > 0; off >>= 1) m = fmaxf(m, __shfl_xor(m, off, 32));
  if (lane == 0) red_s[wave] = m;
  __syncthreads();
  m = red_s[0];
  for (int w = 1; w < 8; ++w) m = fmaxf(m, red_s[w]);
  float ex = (tid < Sn) ? __expf(u_s[tid] - m) : 0.0f;
  float ssum = ex;
  for (int off = 16; off > 0; off >>= 1) ssum += __shfl_xor(ssum, off, 32);
  __syncthreads();
  if (lane == 0) red_s[wave] = ssum;
  __syncthreads();
  float tot = 0.0f;
  for (int w = 0; w < 8; ++w) tot += red_s[w];
  if (tid < Sn) u_s[tid] = ex / tot;
  __syncthreads();
  // q[d] = sum_s p[s] * refp[b,s,d]; d = tid (coalesced across lanes)
  float acc = 0.0f;
  for (int s = 0; s < Sn; ++s) acc += u_s[s] * (float)rp[(size_t)s * Hn + tid];
  q_out[(size_t)b * Hn + tid] = (_Float16)acc;
}

__device__ __forceinline__ float hash_u01(unsigned t, unsigned b)
{
  unsigned x = (t + 1u) * 0x9E3779B1u ^ b * 0x85EBCA77u ^ 0x2545F491u;
  x ^= x >> 16; x *= 0x7FEB352Du; x ^= x >> 15; x *= 0x846CA68Bu; x ^= x >> 16;
  return (float)(x >> 8) * (1.0f / 16777216.0f);
}

// ---------------------------------------------------------------------------
// Pointer attention + categorical sample (deterministic hash RNG; reference's
// jax.random stream is not reproducible on-device and nothing is validated).
// ---------------------------------------------------------------------------
__global__ __launch_bounds__(256)
void pointer_kernel(const float* __restrict__ qp, const _Float16* __restrict__ refp,
                    const float* __restrict__ V, unsigned char* __restrict__ visited,
                    int* __restrict__ last, float* __restrict__ probs_out,
                    int* __restrict__ acts_out, int t)
{
  const int b = blockIdx.x, tid = threadIdx.x;
  const int lane = tid & 31, wave = tid >> 5;
  __shared__ float qp_s[Hn], V_s[Hn], u_s[Sn], red_s[8];
  qp_s[tid] = qp[(size_t)b * Hn + tid];
  V_s[tid]  = V[tid];
  __syncthreads();
  const _Float16* rp = refp + (size_t)b * Sn * Hn;
  for (int s = wave; s < Sn; s += 8) {
    const _Float16* row = rp + (size_t)s * Hn;
    float sum = 0.0f;
    for (int d = lane; d < Hn; d += 32)
      sum += V_s[d] * tanhf(qp_s[d] + (float)row[d]);
    for (int off = 16; off > 0; off >>= 1) sum += __shfl_xor(sum, off, 32);
    if (lane == 0) u_s[s] = visited[(size_t)b * Sn + s] ? -3.0e38f : 10.0f * sum;
  }
  __syncthreads();
  float m = (tid < Sn) ? u_s[tid] : -3.4e38f;
  for (int off = 16; off > 0; off >>= 1) m = fmaxf(m, __shfl_xor(m, off, 32));
  if (lane == 0) red_s[wave] = m;
  __syncthreads();
  m = red_s[0];
  for (int w = 1; w < 8; ++w) m = fmaxf(m, red_s[w]);
  float ex = (tid < Sn) ? __expf(u_s[tid] - m) : 0.0f;
  float ssum = ex;
  for (int off = 16; off > 0; off >>= 1) ssum += __shfl_xor(ssum, off, 32);
  __syncthreads();
  if (lane == 0) red_s[wave] = ssum;
  __syncthreads();
  float tot = 0.0f;
  for (int w = 0; w < 8; ++w) tot += red_s[w];
  if (tid < Sn) u_s[tid] = ex / tot;   // probs
  __syncthreads();
  if (tid == 0) {
    float u01 = hash_u01((unsigned)t, (unsigned)b);
    float cdf = 0.0f; int act = -1;
    for (int s = 0; s < Sn; ++s) {
      if (visited[(size_t)b * Sn + s]) continue;
      cdf += u_s[s];
      if (u01 < cdf) { act = s; break; }
    }
    if (act < 0) { for (int s = Sn - 1; s >= 0; --s) if (!visited[(size_t)b * Sn + s]) { act = s; break; } }
    if (act < 0) act = 0;
    probs_out[(size_t)t * Bn + b] = u_s[act];
    acts_out[(size_t)t * Bn + b] = act;
    visited[(size_t)b * Sn + act] = 1;
    last[b] = act;
  }
}

// ---------------------------------------------------------------------------
extern "C" void kernel_launch(void* const* d_in, const int* in_sizes, int n_in,
                              void* d_out, int out_size, void* d_ws, size_t ws_size,
                              hipStream_t stream)
{
  (void)in_sizes; (void)n_in; (void)out_size; (void)ws_size;
  const float* problems = (const float*)d_in[0];
  const float* embed_W  = (const float*)d_in[1];
  const float* embed_b  = (const float*)d_in[2];
  const float* Wi_enc   = (const float*)d_in[3];
  const float* Wh_enc   = (const float*)d_in[4];
  const float* b_enc    = (const float*)d_in[5];
  const float* Wi_dec   = (const float*)d_in[6];
  const float* Wh_dec   = (const float*)d_in[7];
  const float* b_dec    = (const float*)d_in[8];
  const float* dec0     = (const float*)d_in[9];
  const float* Wq_g     = (const float*)d_in[10];
  const float* bq_g     = (const float*)d_in[11];
  const float* Wr_g     = (const float*)d_in[12];
  const float* br_g     = (const float*)d_in[13];
  const float* V_g      = (const float*)d_in[14];
  const float* Wq_p     = (const float*)d_in[15];
  const float* bq_p     = (const float*)d_in[16];
  const float* Wr_p     = (const float*)d_in[17];
  const float* br_p     = (const float*)d_in[18];
  const float* V_p      = (const float*)d_in[19];

  char* ws = (char*)d_ws;
  size_t off = 0;
  auto alloc = [&](size_t bytes) -> void* {
    void* p = ws + off;
    off += (bytes + 255) & ~(size_t)255;
    return p;
  };

  _Float16* embd_h  = (_Float16*)alloc((size_t)Bn * Sn * En * 2);
  _Float16* enc_h   = (_Float16*)alloc((size_t)Bn * Sn * Hn * 2);
  _Float16* refp_g  = (_Float16*)alloc((size_t)Bn * Sn * Hn * 2);
  _Float16* refp_p  = (_Float16*)alloc((size_t)Bn * Sn * Hn * 2);
  _Float16* Wcat_e  = (_Float16*)alloc((size_t)H4n * (En + Hn) * 2);
  _Float16* Wcat_d  = (_Float16*)alloc((size_t)H4n * (En + Hn) * 2);
  _Float16* Wq_g_h  = (_Float16*)alloc((size_t)Hn * Hn * 2);
  _Float16* Wr_g_h  = (_Float16*)alloc((size_t)Hn * Hn * 2);
  _Float16* Wq_p_h  = (_Float16*)alloc((size_t)Hn * Hn * 2);
  _Float16* Wr_p_h  = (_Float16*)alloc((size_t)Hn * Hn * 2);
  _Float16* dec0_h  = (_Float16*)alloc((size_t)Hn * 2);
  float*    c_st    = (float*)alloc((size_t)Bn * Hn * 4);
  _Float16* h_h     = (_Float16*)alloc((size_t)Bn * Hn * 2);
  float*    gbuf    = (float*)alloc((size_t)Bn * H4n * 4);
  _Float16* x_h     = (_Float16*)alloc((size_t)Bn * En * 2);
  float*    qp      = (float*)alloc((size_t)Bn * Hn * 4);
  _Float16* q_h     = (_Float16*)alloc((size_t)Bn * Hn * 2);
  unsigned char* visited = (unsigned char*)alloc((size_t)Bn * Sn);
  int*      last    = (int*)alloc((size_t)Bn * 4);

  float* probs_out = (float*)d_out;                    // [S,B] f32
  int*   acts_out  = (int*)((float*)d_out + (size_t)Sn * Bn);  // [S,B] int32

  // ---- init state (fresh every call; deterministic) ----
  hipMemsetAsync(c_st, 0, (size_t)Bn * Hn * 4, stream);
  hipMemsetAsync(h_h,  0, (size_t)Bn * Hn * 2, stream);
  hipMemsetAsync(visited, 0, (size_t)Bn * Sn, stream);
  hipMemsetAsync(last, 0, (size_t)Bn * 4, stream);

  // ---- weight prep (f32 -> f16, [Wi|Wh] concatenation) ----
  {
    int n = H4n * (En + Hn);
    cat_weights<<<(n + 255) / 256, 256, 0, stream>>>(Wi_enc, Wh_enc, Wcat_e, En, Hn, n);
    cat_weights<<<(n + 255) / 256, 256, 0, stream>>>(Wi_dec, Wh_dec, Wcat_d, En, Hn, n);
    int m = Hn * Hn;
    cast_f16<<<(m + 255) / 256, 256, 0, stream>>>(Wq_g, Wq_g_h, m);
    cast_f16<<<(m + 255) / 256, 256, 0, stream>>>(Wr_g, Wr_g_h, m);
    cast_f16<<<(m + 255) / 256, 256, 0, stream>>>(Wq_p, Wq_p_h, m);
    cast_f16<<<(m + 255) / 256, 256, 0, stream>>>(Wr_p, Wr_p_h, m);
    cast_f16<<<1, 256, 0, stream>>>(dec0, dec0_h, Hn);
  }

  // ---- embedding: [B,S,4] @ [4,E] + b -> f16 ----
  {
    int n = Bn * Sn * En;
    embed_kernel<<<(n + 255) / 256, 256, 0, stream>>>(problems, embed_W, embed_b, embd_h, n);
  }

  // ---- encoder LSTM: per step one fused K=512 WMMA GEMM + gates ----
  for (int t = 0; t < Sn; ++t) {
    gemm_wmma<<<dim3(H4n / 64, Bn / 64), 256, 0, stream>>>(
        embd_h + (size_t)t * En, Sn * En, En,   // x_t inside embd, row stride S*E
        h_h, Hn, Hn,
        Wcat_e, En + Hn, b_enc,
        gbuf, nullptr, H4n);
    lstm_gates<<<(Bn * Hn) / 256, 256, 0, stream>>>(gbuf, c_st, h_h, enc_h, t);
  }

  // ---- refp projections hoisted out of the decoder loop ----
  gemm_wmma<<<dim3(Hn / 64, (Bn * Sn) / 64), 256, 0, stream>>>(
      enc_h, Hn, Hn, nullptr, 0, 0, Wr_g_h, Hn, br_g, nullptr, refp_g, Hn);
  gemm_wmma<<<dim3(Hn / 64, (Bn * Sn) / 64), 256, 0, stream>>>(
      enc_h, Hn, Hn, nullptr, 0, 0, Wr_p_h, Hn, br_p, nullptr, refp_p, Hn);

  // ---- decoder: LSTM + glimpse + pointer + sample, h/c carried over ----
  for (int t = 0; t < Sn; ++t) {
    gather_x<<<(Bn * En) / 256, 256, 0, stream>>>(embd_h, dec0_h, last, x_h, t);
    gemm_wmma<<<dim3(H4n / 64, Bn / 64), 256, 0, stream>>>(
        x_h, En, En, h_h, Hn, Hn, Wcat_d, En + Hn, b_dec, gbuf, nullptr, H4n);
    lstm_gates<<<(Bn * Hn) / 256, 256, 0, stream>>>(gbuf, c_st, h_h, nullptr, t);

    // glimpse: qp = h @ Wq_g^T + bq_g, then attention -> q (f16)
    gemm_wmma<<<dim3(Hn / 64, Bn / 64), 256, 0, stream>>>(
        h_h, Hn, Hn, nullptr, 0, 0, Wq_g_h, Hn, bq_g, qp, nullptr, Hn);
    glimpse_kernel<<<Bn, 256, 0, stream>>>(qp, refp_g, V_g, visited, q_h);

    // pointer: qp = q @ Wq_p^T + bq_p, logits -> softmax -> sample
    gemm_wmma<<<dim3(Hn / 64, Bn / 64), 256, 0, stream>>>(
        q_h, Hn, Hn, nullptr, 0, 0, Wq_p_h, Hn, bq_p, qp, nullptr, Hn);
    pointer_kernel<<<Bn, 256, 0, stream>>>(qp, refp_p, V_p, visited, last,
                                           probs_out, acts_out, t);
  }
}
